// EncoderLayer_25323127177612
// MI455X (gfx1250) — matrix operand order repaired and checked by reading, not compiled, not run on hardware
//
#include <hip/hip_runtime.h>
#include <hip/hip_bf16.h>

typedef __attribute__((ext_vector_type(16))) __bf16       v16bf;
typedef __attribute__((ext_vector_type(8)))  float        v8f;
typedef __attribute__((ext_vector_type(8)))  unsigned int v8u;
typedef unsigned short ushort_t;

static __device__ __forceinline__ ushort_t f2bf(float f) {
    unsigned u = __builtin_bit_cast(unsigned, f);
    u += 0x7FFFu + ((u >> 16) & 1u);            // round-to-nearest-even
    return (ushort_t)(u >> 16);
}

static __device__ __forceinline__ v8u frag2(uint4 lo, uint4 hi) {
    v8u u;
    u[0] = lo.x; u[1] = lo.y; u[2] = lo.z; u[3] = lo.w;
    u[4] = hi.x; u[5] = hi.y; u[6] = hi.z; u[7] = hi.w;
    return u;
}

static __device__ __forceinline__ v8f wmma_bf16(v8u a, v8u b, v8f c) {
    return __builtin_amdgcn_wmma_f32_16x16x32_bf16(
        false, __builtin_bit_cast(v16bf, a),
        false, __builtin_bit_cast(v16bf, b),
        (short)0, c, false, false);
}

// --- CDNA5 async global->LDS copy (ASYNCcnt-tracked, no VGPR round trip) ---
static __device__ __forceinline__ unsigned lds_off_of(const void* p) {
    return (unsigned)(unsigned long long)
        (__attribute__((address_space(3))) const void*)p;
}
static __device__ __forceinline__ void async_copy_b128(const void* gsrc, void* ldst) {
    asm volatile("global_load_async_to_lds_b128 %0, %1, off"
                 :: "v"(lds_off_of(ldst)), "v"(gsrc) : "memory");
}
static __device__ __forceinline__ void wait_async0() {
    asm volatile("s_wait_asynccnt 0x0" ::: "memory");
}

// ---------------------------------------------------------------- convert
__global__ __launch_bounds__(256) void cvt_f32_bf16(
    const float* __restrict__ src, ushort_t* __restrict__ dst, int n4)
{
    int i = blockIdx.x * 256 + threadIdx.x;
    if (i >= n4) return;
    float4 v = ((const float4*)src)[i];
    ushort4 o;
    o.x = f2bf(v.x); o.y = f2bf(v.y); o.z = f2bf(v.z); o.w = f2bf(v.w);
    ((ushort4*)dst)[i] = o;
}

// ---------------------------------------------------------------- GEMM
// C[M,N] = A[M,K](bf16 rm) x B[K,N](bf16 rm), fp32 accum.
// Macrotile 128x128, BK=32, double-buffered LDS.
// 8 waves in 4(M)x2(N) grid; wave tile 32x64 = 2x4 WMMA subtiles.
// A staged via global_load_async_to_lds_b128; B transposed via register path.
template<bool BIAS, bool RELU, bool OUTBF>
__global__ __launch_bounds__(256) void gemm_bf16_wmma(
    const ushort_t* __restrict__ A, const ushort_t* __restrict__ Bw,
    const float* __restrict__ bias, void* __restrict__ Cout,
    int M, int N, int K)
{
    __shared__ ushort_t As[2][128 * 32];   // [row][k]
    __shared__ ushort_t Bs[2][128 * 32];   // [col][k] (transposed during stage)

    const int blockM = blockIdx.y * 128;
    const int blockN = blockIdx.x * 128;
    const int tid  = threadIdx.x;
    const int lane = tid & 31;
    const int wid  = tid >> 5;
    const int wm = (wid & 3) * 32;
    const int wn = (wid >> 2) * 64;
    const int ml = lane & 15;
    const int kg = lane >> 4;

    // A staging: 512 b128 chunks, 2 per thread. chunk c: row=c>>2, halfword off=(c&3)*8
    const int ac0 = tid * 2;
    // B staging: thread owns k rows {kpair, kpair+1} x 8 cols at bn0
    const int kpair = (tid & 15) * 2;
    const int bn0   = (tid >> 4) * 8;

    const v8f zero = {0.f,0.f,0.f,0.f,0.f,0.f,0.f,0.f};
    v8f acc[2][4];
    #pragma unroll
    for (int mi = 0; mi < 2; ++mi)
        #pragma unroll
        for (int ni = 0; ni < 4; ++ni) acc[mi][ni] = zero;

    auto stageA = [&](int k0, int buf) {
        #pragma unroll
        for (int c = ac0; c < ac0 + 2; ++c) {
            int r = c >> 2, co = (c & 3) * 8;
            async_copy_b128(A + (size_t)(blockM + r) * K + k0 + co,
                            &As[buf][r * 32 + co]);
        }
    };
    auto stageB = [&](int k0, int buf) {
        const ushort_t* bp = Bw + (size_t)(k0 + kpair) * N + blockN + bn0;
        uint4 r0 = *(const uint4*)bp;
        uint4 r1 = *(const uint4*)(bp + N);
        unsigned lo[4] = {r0.x, r0.y, r0.z, r0.w};
        unsigned hi[4] = {r1.x, r1.y, r1.z, r1.w};
        #pragma unroll
        for (int j = 0; j < 8; ++j) {
            unsigned a = (lo[j >> 1] >> ((j & 1) * 16)) & 0xFFFFu;
            unsigned b = (hi[j >> 1] >> ((j & 1) * 16)) & 0xFFFFu;
            *(unsigned*)&Bs[buf][(bn0 + j) * 32 + kpair] = a | (b << 16);
        }
    };

    const int nIter = K >> 5;
    stageA(0, 0);
    stageB(0, 0);

    for (int i = 0; i < nIter; ++i) {
        wait_async0();          // own async A chunks for tile i have landed
        __syncthreads();        // everyone's tile i visible; tile i-1 compute done
        const int cur = i & 1;
        if (i + 1 < nIter) {
            stageA((i + 1) << 5, cur ^ 1);   // in flight under compute
            stageB((i + 1) << 5, cur ^ 1);
        }

        // fragments: each is two contiguous ds_load_b128 per ISA layout
        v8u af[2], bf[4];
        #pragma unroll
        for (int x = 0; x < 2; ++x) {
            const ushort_t* p = &As[cur][(wm + x * 16 + ml) * 32 + 8 * kg];
            af[x] = frag2(*(const uint4*)p, *(const uint4*)(p + 16));
        }
        #pragma unroll
        for (int y = 0; y < 4; ++y) {
            const ushort_t* p = &Bs[cur][(wn + y * 16 + ml) * 32 + 8 * kg];
            bf[y] = frag2(*(const uint4*)p, *(const uint4*)(p + 16));
        }
        #pragma unroll
        for (int x = 0; x < 2; ++x)
            #pragma unroll
            for (int y = 0; y < 4; ++y)
                acc[x][y] = wmma_bf16(af[x], bf[y], acc[x][y]);
    }

    // D layout: lane n=lane&15, mhalf=lane>>4; vgpr r -> m = r + 8*mhalf
    #pragma unroll
    for (int mi = 0; mi < 2; ++mi)
    #pragma unroll
    for (int ni = 0; ni < 4; ++ni) {
        int col = blockN + wn + ni * 16 + ml;
        float bv = BIAS ? bias[col] : 0.0f;
        #pragma unroll
        for (int r = 0; r < 8; ++r) {
            int row = blockM + wm + mi * 16 + r + 8 * kg;
            float val = acc[mi][ni][r] + bv;
            if (RELU) val = fmaxf(val, 0.0f);
            size_t idx = (size_t)row * N + col;
            if (OUTBF) ((ushort_t*)Cout)[idx] = f2bf(val);
            else       ((float*)Cout)[idx]   = val;
        }
    }
}

// ---------------------------------------------------------------- attention
// Flash-style. 128 threads = 4 waves; each wave owns 16 query rows, WG owns 64.
// Q,K,V,ctx are bf16 [B*S, H]; head h occupies columns h*128..h*128+127.
__global__ __launch_bounds__(128) void attn_flash_wmma(
    const ushort_t* __restrict__ Qb, const ushort_t* __restrict__ Kb,
    const ushort_t* __restrict__ Vb, const float* __restrict__ mask,
    ushort_t* __restrict__ ctx)
{
    constexpr int S = 2048, H = 1024, D = 128;
    __shared__ ushort_t Vt[D * 32];       // V tile transposed: [col][key]
    __shared__ ushort_t Pl[4][16 * 32];   // per-wave P (probabilities) bounce

    const int b = blockIdx.z, h = blockIdx.y;
    const int tid = threadIdx.x, lane = tid & 31, wid = tid >> 5;
    const int qbase = blockIdx.x * 64 + wid * 16;
    const int ml = lane & 15, kg = lane >> 4;

    // Q A-fragments: 16 rows x 128 deep -> 4 frags (2x b128 each), kept in VGPRs
    v8u aq[4];
    {
        const ushort_t* qrow = Qb + ((size_t)(b * S + qbase + ml)) * H + h * D;
        #pragma unroll
        for (int c = 0; c < 4; ++c) {
            const ushort_t* p = qrow + c * 32 + 16 * kg;
            aq[c] = frag2(*(const uint4*)p, *(const uint4*)(p + 8));
        }
    }

    const v8f zero = {0.f,0.f,0.f,0.f,0.f,0.f,0.f,0.f};
    v8f o[8];
    #pragma unroll
    for (int j = 0; j < 8; ++j) o[j] = zero;
    float rmax[8], rsum[8];
    #pragma unroll
    for (int r = 0; r < 8; ++r) { rmax[r] = -3.0e38f; rsum[r] = 0.f; }

    const float sc = 11.313708498984761f;  // sqrt(128) -- faithful to reference bug
    const ushort_t* kp = Kb + ((size_t)(b * S)) * H + h * D;

    for (int kb = 0; kb < S; kb += 32) {
        { // cooperative stage of V[kb..kb+31][:] transposed
            int key = tid >> 2, c0 = (tid & 3) * 32;
            const uint4* vp = (const uint4*)(Vb + ((size_t)(b*S + kb + key)) * H + h * D + c0);
            #pragma unroll
            for (int q = 0; q < 4; ++q) {
                uint4 dv = vp[q];
                unsigned w[4] = {dv.x, dv.y, dv.z, dv.w};
                #pragma unroll
                for (int wi = 0; wi < 4; ++wi) {
                    int cc = c0 + q*8 + wi*2;
                    Vt[(cc + 0) * 32 + key] = (ushort_t)(w[wi] & 0xFFFFu);
                    Vt[(cc + 1) * 32 + key] = (ushort_t)(w[wi] >> 16);
                }
            }
        }
        __syncthreads();

        // scores: 16 q x 32 keys as two 16x16 accumulators
        v8f s0 = zero, s1 = zero;
        #pragma unroll
        for (int c = 0; c < 4; ++c) {
            const ushort_t* p0 = kp + (size_t)(kb      + ml) * H + c * 32 + 16 * kg;
            const ushort_t* p1 = kp + (size_t)(kb + 16 + ml) * H + c * 32 + 16 * kg;
            v8u bk0 = frag2(*(const uint4*)p0, *(const uint4*)(p0 + 8));
            v8u bk1 = frag2(*(const uint4*)p1, *(const uint4*)(p1 + 8));
            s0 = wmma_bf16(aq[c], bk0, s0);
            s1 = wmma_bf16(aq[c], bk1, s1);
        }

        float mk0 = mask[b * S + kb + ml]      * -1.0e9f;
        float mk1 = mask[b * S + kb + 16 + ml] * -1.0e9f;

        #pragma unroll
        for (int r = 0; r < 8; ++r) {
            float e0 = s0[r] * sc + mk0;
            float e1 = s1[r] * sc + mk1;
            float t = fmaxf(e0, e1);
            #pragma unroll
            for (int off = 1; off < 16; off <<= 1)          // stays within 16-lane half
                t = fmaxf(t, __shfl_xor(t, off, 32));
            float nm    = fmaxf(rmax[r], t);
            float alpha = __expf(rmax[r] - nm);
            float p0 = __expf(e0 - nm);
            float p1 = __expf(e1 - nm);
            float ps = p0 + p1;
            #pragma unroll
            for (int off = 1; off < 16; off <<= 1)
                ps += __shfl_xor(ps, off, 32);
            rsum[r] = rsum[r] * alpha + ps;
            rmax[r] = nm;
            #pragma unroll
            for (int j = 0; j < 8; ++j) o[j][r] *= alpha;
            int m = r + 8 * kg;                              // D-layout row
            Pl[wid][m * 32 + ml]      = f2bf(p0);
            Pl[wid][m * 32 + 16 + ml] = f2bf(p1);
        }

        asm volatile("s_wait_dscnt 0x0" ::: "memory");       // wave-local LDS RAW

        const ushort_t* pp = &Pl[wid][ml * 32 + 16 * kg];
        v8u pa = frag2(*(const uint4*)pp, *(const uint4*)(pp + 8));
        #pragma unroll
        for (int j = 0; j < 8; ++j) {                        // P(16x32) x V(32x16) per d-chunk
            const ushort_t* vp2 = &Vt[(j * 16 + ml) * 32 + 16 * kg];
            v8u bv = frag2(*(const uint4*)vp2, *(const uint4*)(vp2 + 8));
            o[j] = wmma_bf16(pa, bv, o[j]);
        }
        __syncthreads();   // protect Vt before next tile staging
    }

    #pragma unroll
    for (int r = 0; r < 8; ++r) {
        float inv = 1.0f / rsum[r];
        int m = r + 8 * kg;
        ushort_t* crow = ctx + ((size_t)(b * S + qbase + m)) * H + h * D;
        #pragma unroll
        for (int j = 0; j < 8; ++j)
            crow[j * 16 + ml] = f2bf(o[j][r] * inv);
    }
}

// ---------------------------------------------------------------- residual + LayerNorm
__global__ __launch_bounds__(256) void ln_residual(
    const float* __restrict__ xin, const float* __restrict__ res,
    const float* __restrict__ gamma, const float* __restrict__ beta,
    float* __restrict__ outf, ushort_t* __restrict__ outb)
{
    constexpr int H = 1024;
    const int lane = threadIdx.x & 31, wid = threadIdx.x >> 5;
    const size_t row = (size_t)blockIdx.x * 8 + wid;
    const float4* xp = (const float4*)(xin + row * H);
    const float4* rp = (const float4*)(res + row * H);
    float4 v[8];
    float sum = 0.f, sq = 0.f;
    #pragma unroll
    for (int t = 0; t < 8; ++t) {
        int idx = t * 32 + lane;
        float4 a = xp[idx], b = rp[idx];
        float4 c; c.x = a.x + b.x; c.y = a.y + b.y; c.z = a.z + b.z; c.w = a.w + b.w;
        v[t] = c;
        sum += c.x + c.y + c.z + c.w;
        sq  += c.x*c.x + c.y*c.y + c.z*c.z + c.w*c.w;
    }
    #pragma unroll
    for (int off = 1; off < 32; off <<= 1) {
        sum += __shfl_xor(sum, off, 32);
        sq  += __shfl_xor(sq,  off, 32);
    }
    float mean = sum * (1.0f / H);
    float var  = sq  * (1.0f / H) - mean * mean;
    float rstd = rsqrtf(var + 1e-6f);
    const float4* gp = (const float4*)gamma;
    const float4* bp = (const float4*)beta;
    #pragma unroll
    for (int t = 0; t < 8; ++t) {
        int idx = t * 32 + lane;
        float4 g = gp[idx], bb = bp[idx];
        float4 c = v[t], ov;
        ov.x = (c.x - mean) * rstd * g.x + bb.x;
        ov.y = (c.y - mean) * rstd * g.y + bb.y;
        ov.z = (c.z - mean) * rstd * g.z + bb.z;
        ov.w = (c.w - mean) * rstd * g.w + bb.w;
        ((float4*)(outf + row * H))[idx] = ov;
        if (outb) {
            ushort4 ob;
            ob.x = f2bf(ov.x); ob.y = f2bf(ov.y); ob.z = f2bf(ov.z); ob.w = f2bf(ov.w);
            ((ushort4*)(outb + row * H))[idx] = ob;
        }
    }
}

// ---------------------------------------------------------------- launch
extern "C" void kernel_launch(void* const* d_in, const int* in_sizes, int n_in,
                              void* d_out, int out_size, void* d_ws, size_t ws_size,
                              hipStream_t stream)
{
    (void)in_sizes; (void)n_in; (void)out_size; (void)ws_size;
    constexpr int Bz = 4, S = 2048, H = 1024, F = 4096, NHh = 8;
    constexpr size_t BS = (size_t)Bz * S;

    const float* x     = (const float*)d_in[0];
    const float* mask  = (const float*)d_in[1];
    const float* Wq    = (const float*)d_in[2];
    const float* Wk    = (const float*)d_in[3];
    const float* Wv    = (const float*)d_in[4];
    const float* Wo    = (const float*)d_in[5];
    const float* W1    = (const float*)d_in[6];
    const float* b1    = (const float*)d_in[7];
    const float* W2    = (const float*)d_in[8];
    const float* b2    = (const float*)d_in[9];
    const float* gamma = (const float*)d_in[10];
    const float* beta  = (const float*)d_in[11];
    float* out = (float*)d_out;

    char* ws = (char*)d_ws;
    size_t off = 0;
    auto alloc = [&](size_t bytes) { char* p = ws + off; off += bytes; return p; };

    ushort_t* xb  = (ushort_t*)alloc(BS * H * 2);
    ushort_t* wqb = (ushort_t*)alloc((size_t)H * H * 2);
    ushort_t* wkb = (ushort_t*)alloc((size_t)H * H * 2);
    ushort_t* wvb = (ushort_t*)alloc((size_t)H * H * 2);
    ushort_t* wob = (ushort_t*)alloc((size_t)H * H * 2);
    ushort_t* w1b = (ushort_t*)alloc((size_t)H * F * 2);
    ushort_t* w2b = (ushort_t*)alloc((size_t)F * H * 2);
    char* qkvc = alloc(4 * BS * H * 2);            // Q,K,V,ctx -> later reused as FFN hidden
    ushort_t* Qb   = (ushort_t*)qkvc;
    ushort_t* Kb   = Qb + BS * H;
    ushort_t* Vb   = Kb + BS * H;
    ushort_t* ctxb = Vb + BS * H;
    ushort_t* hb   = (ushort_t*)qkvc;              // BS*F*2 == 4*BS*H*2 exactly
    float* attn_out = (float*)alloc(BS * H * 4);   // reused as FFN output
    float* ffn_out  = attn_out;
    float* x1       = (float*)alloc(BS * H * 4);
    ushort_t* x1b   = (ushort_t*)alloc(BS * H * 2);

    auto cvt = [&](const float* s, ushort_t* d, size_t n) {
        int n4 = (int)(n / 4);
        cvt_f32_bf16<<<(n4 + 255) / 256, 256, 0, stream>>>(s, d, n4);
    };
    cvt(x,  xb,  BS * H);
    cvt(Wq, wqb, (size_t)H * H);
    cvt(Wk, wkb, (size_t)H * H);
    cvt(Wv, wvb, (size_t)H * H);
    cvt(Wo, wob, (size_t)H * H);
    cvt(W1, w1b, (size_t)H * F);
    cvt(W2, w2b, (size_t)F * H);

    dim3 blk(256);
    dim3 gHH(H / 128, (unsigned)(BS / 128));       // M=8192, N=1024
    gemm_bf16_wmma<false,false,true ><<<gHH, blk, 0, stream>>>(xb, wqb, nullptr, Qb, (int)BS, H, H);
    gemm_bf16_wmma<false,false,true ><<<gHH, blk, 0, stream>>>(xb, wkb, nullptr, Kb, (int)BS, H, H);
    gemm_bf16_wmma<false,false,true ><<<gHH, blk, 0, stream>>>(xb, wvb, nullptr, Vb, (int)BS, H, H);

    dim3 ga(S / 64, NHh, Bz);
    attn_flash_wmma<<<ga, 128, 0, stream>>>(Qb, Kb, Vb, mask, ctxb);

    gemm_bf16_wmma<false,false,false><<<gHH, blk, 0, stream>>>(ctxb, wob, nullptr, attn_out, (int)BS, H, H);

    ln_residual<<<(unsigned)(BS / 8), 256, 0, stream>>>(x, attn_out, gamma, beta, x1, x1b);

    dim3 gHF(F / 128, (unsigned)(BS / 128));       // M=8192, N=4096
    gemm_bf16_wmma<true, true, true ><<<gHF, blk, 0, stream>>>(x1b, w1b, b1, hb, (int)BS, F, H);
    gemm_bf16_wmma<true, false,false><<<gHH, blk, 0, stream>>>(hb,  w2b, b2, ffn_out, (int)BS, H, F);

    ln_residual<<<(unsigned)(BS / 8), 256, 0, stream>>>(x1, ffn_out, gamma, beta, out, nullptr);
}
